// EntityExtractor_29944511987733
// MI455X (gfx1250) — matrix operand order repaired and checked by reading, not compiled, not run on hardware
//
#include <hip/hip_runtime.h>
#include <hip/hip_bf16.h>

typedef __attribute__((ext_vector_type(16))) _Float16 v16h;
typedef __attribute__((ext_vector_type(8)))  float    v8f;

#define NCELLS 4096   // 64 batch * 64 cells
// ---------------------------------------------------------------------------
// Kernel 1: per-cell conv1 (3->16ch, 8x8, stride4, pad2) as implicit GEMM.
// One workgroup (128 thr = 4 waves) per cell. M=144 pix (9 tiles), N=16, K=192.
// Input cell staged in LDS as padded 52x52x3 f16 (x read exactly once).
// ---------------------------------------------------------------------------
__global__ __launch_bounds__(128) void conv1_wmma(
    const float* __restrict__ x, const float* __restrict__ w1,
    const float* __restrict__ b1, _Float16* __restrict__ out1)
{
    __shared__ _Float16 s_in[3 * 52 * 52];

    const int cell = blockIdx.x;
    const int b  = cell >> 6;
    const int gx = (cell >> 3) & 7;
    const int gy = cell & 7;
    const float* xb = x + (size_t)b * 3 * 384 * 384;

    // Stage padded cell (pad=2 each side) into LDS, fp32 -> f16.
    for (int i = threadIdx.x; i < 3 * 52 * 52; i += 128) {
        int c  = i / (52 * 52);
        int r  = i % (52 * 52);
        int iy = r / 52 - 2;
        int ix = r % 52 - 2;
        float v = 0.0f;
        if (iy >= 0 && iy < 48 && ix >= 0 && ix < 48)
            v = xb[((size_t)c * 384 + gx * 48 + iy) * 384 + gy * 48 + ix];
        s_in[i] = (_Float16)v;
    }
    __syncthreads();

    const int wave = threadIdx.x >> 5;
    const int lane = threadIdx.x & 31;
    const int hi   = (lane >= 16);
    const int mrow = lane & 15;        // A: M row within tile
    const int nch  = lane & 15;        // B/D: output channel
    const int kA   = hi ? 8 : 0;       // A-fragment K base (16-bit layout)
    const int kB   = hi ? 8 : 0;       // B-fragment K base
    const float bv = b1[nch];

    for (int t = wave; t < 9; t += 4) {          // 9 M-tiles over 4 waves
        const int p  = t * 16 + mrow;            // output pixel 0..143
        const int oy = p / 12, ox = p % 12;
        v8f acc = {};
        for (int ks = 0; ks < 6; ++ks) {         // K = 192 = 6 * 32
            const int kbase = ks * 32;
            v16h a, bm;
#pragma unroll
            for (int e = 0; e < 16; ++e) {
                int K  = kbase + kA + (e < 8 ? e : e + 8);
                int c  = K >> 6, rem = K & 63;
                int kh = rem >> 3, kw = rem & 7;
                a[e] = s_in[c * 52 * 52 + (oy * 4 + kh) * 52 + (ox * 4 + kw)];
            }
#pragma unroll
            for (int e = 0; e < 16; ++e) {
                int K = kbase + kB + (e < 8 ? e : e + 8);          // w1: [16][3*8*8] row-major
                bm[e] = (_Float16)w1[nch * 192 + K];
            }
            acc = __builtin_amdgcn_wmma_f32_16x16x32_f16(
                false, a, false, bm, (short)0, acc, false, false);
        }
        // D layout: VGPR r -> M = r + 8*hi, N = lane&15. ReLU + bias, f16 out.
#pragma unroll
        for (int r = 0; r < 8; ++r) {
            int pr = t * 16 + r + (hi ? 8 : 0);
            float v = acc[r] + bv;
            v = v > 0.0f ? v : 0.0f;
            out1[((size_t)cell * 144 + pr) * 16 + nch] = (_Float16)v;
        }
    }
}

// ---------------------------------------------------------------------------
// Kernel 2: per-cell conv2 (16->32ch, 4x4, stride4, pad1). One wave per cell.
// M = 9 out pixels (padded to 16), N = 32 (2 tiles), K = 256 (8 steps).
// Produces flattened features h2[cell][288] with ref order feat = ch*9 + pix.
// ---------------------------------------------------------------------------
__global__ __launch_bounds__(128) void conv2_wmma(
    const _Float16* __restrict__ out1, const float* __restrict__ w2,
    const float* __restrict__ b2, _Float16* __restrict__ h2)
{
    const int wave = threadIdx.x >> 5;
    const int lane = threadIdx.x & 31;
    const int cell = blockIdx.x * 4 + wave;
    const int hi   = (lane >= 16);
    const int mrow = lane & 15;        // out pixel (valid < 9)
    const int oy   = mrow / 3, ox = mrow % 3;
    const int kA   = hi ? 8 : 0;
    const int kB   = hi ? 8 : 0;
    const int nloc = lane & 15;
    const _Float16* in = out1 + (size_t)cell * 144 * 16;

    for (int nt = 0; nt < 2; ++nt) {
        const int n = nt * 16 + nloc;            // out channel 0..31
        v8f acc = {};
        for (int ks = 0; ks < 8; ++ks) {         // K = 256 = 8 * 32
            const int kbase = ks * 32;
            v16h a, bm;
#pragma unroll
            for (int e = 0; e < 16; ++e) {
                int K  = kbase + kA + (e < 8 ? e : e + 8);
                int c  = K >> 4;                 // in channel
                int kh = (K >> 2) & 3, kw = K & 3;
                int iy = oy * 4 - 1 + kh;        // pad = 1
                int ix = ox * 4 - 1 + kw;
                _Float16 v = (_Float16)0.0f;
                if (mrow < 9 && iy >= 0 && iy < 12 && ix >= 0 && ix < 12)
                    v = in[(iy * 12 + ix) * 16 + c];
                a[e] = v;
            }
#pragma unroll
            for (int e = 0; e < 16; ++e) {
                int K = kbase + kB + (e < 8 ? e : e + 8);          // w2: [32][16*4*4] row-major
                bm[e] = (_Float16)w2[n * 256 + K];
            }
            acc = __builtin_amdgcn_wmma_f32_16x16x32_f16(
                false, a, false, bm, (short)0, acc, false, false);
        }
        const float bv = b2[n];
#pragma unroll
        for (int r = 0; r < 8; ++r) {
            int p = r + (hi ? 8 : 0);
            if (p < 9) {
                float v = acc[r] + bv;
                v = v > 0.0f ? v : 0.0f;
                h2[(size_t)cell * 288 + n * 9 + p] = (_Float16)v;
            }
        }
    }
}

// ---------------------------------------------------------------------------
// Kernel 3: fc1 GEMM [4096,288] x [288,256]^T + bias, ReLU -> h3 f16.
// 256 M-tiles x 16 N-tiles = 4096 wave-tiles; 8 waves/block, 512 blocks.
// ---------------------------------------------------------------------------
__global__ __launch_bounds__(256) void fc1_wmma(
    const _Float16* __restrict__ h2, const float* __restrict__ fw,
    const float* __restrict__ fb, _Float16* __restrict__ h3)
{
    const int wave = threadIdx.x >> 5;
    const int lane = threadIdx.x & 31;
    const int task = blockIdx.x * 8 + wave;
    const int mt = task >> 4;                    // 0..255
    const int nt = task & 15;                    // 0..15
    const int hi = (lane >= 16);
    const int kA = hi ? 8 : 0;
    const int kB = hi ? 8 : 0;
    const int arow = mt * 16 + (lane & 15);
    const int ncol = nt * 16 + (lane & 15);

    v8f acc = {};
    for (int ks = 0; ks < 9; ++ks) {             // K = 288 = 9 * 32
        const int kbase = ks * 32;
        v16h a, bm;
#pragma unroll
        for (int e = 0; e < 16; ++e) {
            int K = kbase + kA + (e < 8 ? e : e + 8);
            a[e] = h2[(size_t)arow * 288 + K];
        }
#pragma unroll
        for (int e = 0; e < 16; ++e) {
            int K = kbase + kB + (e < 8 ? e : e + 8);              // fw: [256 out][288 in]
            bm[e] = (_Float16)fw[(size_t)ncol * 288 + K];
        }
        acc = __builtin_amdgcn_wmma_f32_16x16x32_f16(
            false, a, false, bm, (short)0, acc, false, false);
    }
    const float bv = fb[ncol];
#pragma unroll
    for (int r = 0; r < 8; ++r) {
        int row = mt * 16 + r + (hi ? 8 : 0);
        float v = acc[r] + bv;
        v = v > 0.0f ? v : 0.0f;
        h3[(size_t)row * 256 + ncol] = (_Float16)v;
    }
}

// ---------------------------------------------------------------------------
// Kernel 4: fc2 (256->4) + softmax, scalar per row (N=4 is WMMA-hostile).
// Output: 3 planes of 4096 probs, concatenated (ref returns feats[:,:,0..2]).
// ---------------------------------------------------------------------------
__global__ __launch_bounds__(256) void fc2_softmax(
    const _Float16* __restrict__ h3, const float* __restrict__ fw,
    const float* __restrict__ fb, float* __restrict__ out)
{
    const int row = blockIdx.x * 256 + threadIdx.x;
    if (row >= NCELLS) return;
    float acc0 = fb[0], acc1 = fb[1], acc2 = fb[2], acc3 = fb[3];
    const _Float16* hr = h3 + (size_t)row * 256;
    for (int k = 0; k < 256; ++k) {
        float hv = (float)hr[k];
        acc0 += hv * fw[0 * 256 + k];
        acc1 += hv * fw[1 * 256 + k];
        acc2 += hv * fw[2 * 256 + k];
        acc3 += hv * fw[3 * 256 + k];
    }
    float m = fmaxf(fmaxf(acc0, acc1), fmaxf(acc2, acc3));
    float e0 = __expf(acc0 - m), e1 = __expf(acc1 - m);
    float e2 = __expf(acc2 - m), e3 = __expf(acc3 - m);
    float inv = 1.0f / (e0 + e1 + e2 + e3);
    out[0 * NCELLS + row] = e0 * inv;
    out[1 * NCELLS + row] = e1 * inv;
    out[2 * NCELLS + row] = e2 * inv;
}

// ---------------------------------------------------------------------------
extern "C" void kernel_launch(void* const* d_in, const int* in_sizes, int n_in,
                              void* d_out, int out_size, void* d_ws, size_t ws_size,
                              hipStream_t stream) {
    const float* x   = (const float*)d_in[0];
    const float* w1  = (const float*)d_in[1];
    const float* b1  = (const float*)d_in[2];
    const float* w2  = (const float*)d_in[3];
    const float* b2  = (const float*)d_in[4];
    const float* fw1 = (const float*)d_in[5];
    const float* fb1 = (const float*)d_in[6];
    const float* fw2 = (const float*)d_in[7];
    const float* fb2 = (const float*)d_in[8];
    float* out = (float*)d_out;

    // Workspace layout (f16 activations; all fit comfortably in the 192MB L2):
    //   out1: 4096*144*16 halfs (18.9 MB)
    //   h2  : 4096*288   halfs ( 2.4 MB)
    //   h3  : 4096*256   halfs ( 2.0 MB)
    char* ws = (char*)d_ws;
    _Float16* out1 = (_Float16*)(ws);
    _Float16* h2   = (_Float16*)(ws + (size_t)NCELLS * 144 * 16 * 2);
    _Float16* h3   = (_Float16*)(ws + (size_t)NCELLS * 144 * 16 * 2
                                    + (size_t)NCELLS * 288 * 2);

    conv1_wmma <<<NCELLS,      128, 0, stream>>>(x, w1, b1, out1);
    conv2_wmma <<<NCELLS / 4,  128, 0, stream>>>(out1, w2, b2, h2);
    fc1_wmma   <<<4096 / 8,    256, 0, stream>>>(h2, fw1, fb1, h3);
    fc2_softmax<<<NCELLS / 256, 256, 0, stream>>>(h3, fw2, fb2, out);
}
